// WanLynxIPCrossAttention_14010183319945
// MI455X (gfx1250) — compile-verified, matrix operand order checked by reading, not run on hardware
//
#include <hip/hip_runtime.h>
#include <hip/hip_bf16.h>

// ---------------------------------------------------------------------------
// WanLynx IP cross-attention for MI455X (gfx1250, wave32, WMMA bf16)
//   D=5120, heads=40, head_dim=128, s=20480, Lk=272 (=256 ip + 16 registers)
// ---------------------------------------------------------------------------

#define D_      5120
#define S_      20480
#define HEADS_  40
#define HD_     128
#define LIP_    256
#define LK_     272            // 17 * 16
#define LKP_    288            // 9 * 32 (padded PV contraction)
#define EPS_    1e-6f

#define KSTR_   144            // bf16/row for K LDS (128+16 pad; 72 dw -> bank rot 8)
#define VSTR_   296            // bf16/col for transposed V LDS (288+8 pad)
#define SSTR_   292            // f32/row score LDS (288+4 pad)
#define PSTR_   292            // bf16/row P LDS

typedef __attribute__((ext_vector_type(16))) __bf16 v16bf;
typedef __attribute__((ext_vector_type(8)))  float  v8f;

union Frag {
    v16bf bf;
    unsigned int   u[8];
    unsigned short us[16];
};

// fast float->bf16 (round-half-up): 1 add per scalar
__device__ __forceinline__ unsigned short f2bf(float f) {
    return (unsigned short)((__float_as_uint(f) + 0x8000u) >> 16);
}
// pack two floats to packed bf16 pair: 2 adds + 1 v_perm_b32
__device__ __forceinline__ unsigned pack_bf2(float a, float b) {
    unsigned ua = __float_as_uint(a) + 0x8000u;
    unsigned ub = __float_as_uint(b) + 0x8000u;
    return __builtin_amdgcn_perm(ub, ua, 0x07060302u);
}

// ISA 16-bit A/B striping (16x32): VGPR v of lane-group g holds K {kb, kb+1},
// kb = 2*(v&3) + 8*g + 16*(v>>2). v-quads {0,1},{2,3},{4,5},{6,7} map to the
// contiguous element runs 8g+{0..3}, 8g+{4..7}, 8g+16+{0..3}, 8g+16+{4..7}.
__device__ __forceinline__ int kbase(int v, int g) {
    return 2 * (v & 3) + 8 * g + 16 * (v >> 2);
}
__device__ __forceinline__ int kquad(int vq, int g) {   // vq = v/2
    return 4 * (vq & 1) + 8 * g + 16 * (vq >> 1);
}

// build one 16x32 bf16 fragment from a row-major fp32 row: 4 x b128 loads
__device__ __forceinline__ void frag_from_f32row(Frag& f, const float* row, int g) {
    const float4 q0 = *(const float4*)(row + 8 * g);
    const float4 q1 = *(const float4*)(row + 8 * g + 4);
    const float4 q2 = *(const float4*)(row + 8 * g + 16);
    const float4 q3 = *(const float4*)(row + 8 * g + 20);
    f.u[0] = pack_bf2(q0.x, q0.y);  f.u[1] = pack_bf2(q0.z, q0.w);
    f.u[2] = pack_bf2(q1.x, q1.y);  f.u[3] = pack_bf2(q1.z, q1.w);
    f.u[4] = pack_bf2(q2.x, q2.y);  f.u[5] = pack_bf2(q2.z, q2.w);
    f.u[6] = pack_bf2(q3.x, q3.y);  f.u[7] = pack_bf2(q3.z, q3.w);
}

// build one 16x32 bf16 fragment from contiguous bf16 LDS: 4 x ds_load_b64
__device__ __forceinline__ void frag_from_bf16lds(Frag& f, const unsigned short* base,
                                                  int g) {
#pragma unroll
    for (int vq = 0; vq < 4; ++vq) {
        const uint2 d = *(const uint2*)(base + kquad(vq, g));
        f.u[2 * vq]     = d.x;
        f.u[2 * vq + 1] = d.y;
    }
}

// ---------------------------------------------------------------------------
// Kernel 1: K = ip @ Wk^T (+bk), V = ip @ Wv^T (+bv); raw fp32 to workspace.
// grid (40, 17), block 256 (8 waves). Wave w: cols [(bx*8+w)*16, +16),
// rows [my*16, +16). Contraction over D in 32-wide bf16 WMMA steps.
// ---------------------------------------------------------------------------
__global__ void kv_proj_kernel(const float* __restrict__ ipx,
                               const float* __restrict__ regs,
                               const float* __restrict__ Wk,
                               const float* __restrict__ bk,
                               const float* __restrict__ Wv,
                               const float* __restrict__ bv,
                               float* __restrict__ kraw,
                               float* __restrict__ vraw) {
    const int tid = threadIdx.x;
    const int l   = tid & 31;
    const int w   = tid >> 5;
    const int g   = l >> 4;
    const int n16 = l & 15;

    const int my = blockIdx.y;
    const int c0 = (blockIdx.x * 8 + w) * 16;

    // A: ip row for this lane (ragged concat of ip_x and registers)
    const int arow = my * 16 + n16;
    const float* asrc = (arow < LIP_) ? (ipx + (size_t)arow * D_)
                                      : (regs + (size_t)(arow - LIP_) * D_);
    const float* wkrow = Wk + (size_t)(c0 + n16) * D_;
    const float* wvrow = Wv + (size_t)(c0 + n16) * D_;

    v8f ak = {}, av = {};
    for (int j0 = 0; j0 < D_; j0 += 32) {
        Frag a, bK, bV;
        frag_from_f32row(a,  asrc  + j0, g);
        frag_from_f32row(bK, wkrow + j0, g);
        frag_from_f32row(bV, wvrow + j0, g);
        ak = __builtin_amdgcn_wmma_f32_16x16x32_bf16(false, a.bf, false, bK.bf,
                                                     (short)0, ak, false, false);
        av = __builtin_amdgcn_wmma_f32_16x16x32_bf16(false, a.bf, false, bV.bf,
                                                     (short)0, av, false, false);
    }

    const int col = c0 + n16;
#pragma unroll
    for (int r = 0; r < 8; ++r) {
        const int row = my * 16 + r + 8 * g;
        kraw[(size_t)row * D_ + col] = ak[r] + bk[col];
        vraw[(size_t)row * D_ + col] = av[r] + bv[col];
    }
}

// ---------------------------------------------------------------------------
// Kernel 2: in-place WanRMSNorm on K rows: k *= rsqrt(mean(k^2)+eps) * w
// ---------------------------------------------------------------------------
__global__ void rmsnorm_k_kernel(float* __restrict__ kraw,
                                 const float* __restrict__ wgt) {
    const int row = blockIdx.x;
    const int tid = threadIdx.x;
    float* kr = kraw + (size_t)row * D_;

    __shared__ float red[256];
    float ss = 0.f;
    for (int c = tid; c < D_; c += 256) {
        const float x = kr[c];
        ss += x * x;
    }
    red[tid] = ss;
    __syncthreads();
    for (int off = 128; off > 0; off >>= 1) {
        if (tid < off) red[tid] += red[tid + off];
        __syncthreads();
    }
    const float scale = rsqrtf(red[0] * (1.0f / D_) + EPS_);
    for (int c = tid; c < D_; c += 256) {
        kr[c] = kr[c] * scale * wgt[c];
    }
}

// ---------------------------------------------------------------------------
// Kernel 3: attention. grid (160, 40), block 256 (8 waves, 16 q-rows/wave).
// LDS layout (bytes):
//   [0, 78336)        : K_h bf16 [272][144]  -- later ALIASED by P bf16 [128][292]
//   [78336, 154112)   : V_h^T bf16 [128 cols][296] (k-major, pad k zero)
//   [154112, 303616)  : scores f32, per-wave [16][292]
//   [303616, 304128)  : per-row 1/sum [128]
// ---------------------------------------------------------------------------
#define SMEM_KP_OFF 0
#define SMEM_KP_SZ  (LK_ * KSTR_ * 2)                    // 78336 (> 128*292*2)
#define SMEM_V_OFF  (SMEM_KP_OFF + SMEM_KP_SZ)           // 78336
#define SMEM_S_OFF  (SMEM_V_OFF + HD_ * VSTR_ * 2)       // 154112
#define SMEM_I_OFF  (SMEM_S_OFF + 8 * 16 * SSTR_ * 4)    // 303616
#define SMEM_TOTAL  (SMEM_I_OFF + 128 * 4)               // 304128 bytes

__global__ void attn_kernel(const float* __restrict__ q,
                            const float* __restrict__ knorm,
                            const float* __restrict__ vraw,
                            float* __restrict__ out) {
    extern __shared__ char smem[];
    unsigned short* kb = (unsigned short*)(smem + SMEM_KP_OFF);  // phase A
    unsigned short* pb = (unsigned short*)(smem + SMEM_KP_OFF);  // phase B alias
    unsigned short* vb = (unsigned short*)(smem + SMEM_V_OFF);   // transposed
    float*          sb = (float*)(smem + SMEM_S_OFF);
    float*          iv = (float*)(smem + SMEM_I_OFF);

    const int tid = threadIdx.x;
    const int l   = tid & 31;
    const int w   = tid >> 5;
    const int g   = l >> 4;
    const int n16 = l & 15;
    const int h   = blockIdx.y;

    // ---- stage K_h (row-major) and V_h (transposed, k-major) as bf16 ----
    const float* kh = knorm + (size_t)h * HD_;
    const float* vh = vraw  + (size_t)h * HD_;
    for (int i = tid; i < LK_ * HD_; i += 256) {
        const int r = i >> 7, c = i & 127;
        kb[r * KSTR_ + c] = f2bf(kh[(size_t)r * D_ + c]);
    }
    for (int i = tid; i < LKP_ * HD_; i += 256) {
        const int k = i >> 7, c = i & 127;          // c fastest: coalesced reads
        vb[c * VSTR_ + k] = (k < LK_) ? f2bf(vh[(size_t)k * D_ + c])
                                      : (unsigned short)0;
    }
    __syncthreads();

    // ---- Q fragments: 4 k-steps of 16x32 bf16, direct from global ----
    const int q0   = blockIdx.x * 128 + w * 16;
    const int qrow = q0 + n16;                // A-matrix lane row (m = lane%16)
    const float* qp = q + (size_t)qrow * D_ + h * HD_;
    Frag aq[4];
#pragma unroll
    for (int ks = 0; ks < 4; ++ks)
        frag_from_f32row(aq[ks], qp + ks * 32, g);

    // ---- scores: 17 tiles of 16 keys, S = (Q K^T) * d^-0.5 -> LDS f32 ----
    float* swave = sb + w * 16 * SSTR_;
    const float scl = 0.08838834764831845f;   // 1/sqrt(128)
    for (int t = 0; t < 17; ++t) {
        const int nk0 = t * 16;
        const unsigned short* krow = kb + (nk0 + n16) * KSTR_;  // lane col n
        v8f acc = {};
#pragma unroll
        for (int ks = 0; ks < 4; ++ks) {
            Frag bK;
            frag_from_bf16lds(bK, krow + ks * 32, g);
            acc = __builtin_amdgcn_wmma_f32_16x16x32_bf16(false, aq[ks].bf, false,
                                                          bK.bf, (short)0, acc,
                                                          false, false);
        }
#pragma unroll
        for (int r = 0; r < 8; ++r)
            swave[(r + 8 * g) * SSTR_ + nk0 + n16] = acc[r] * scl;
    }
    __syncthreads();   // K buffer dead past this point; P may alias it

    // ---- softmax: 2 lanes per row; write exp() as bf16 P + 1/sum ----
    {
        const int row = l >> 1, half = l & 1;
        const float* sr = swave + row * SSTR_;
        unsigned short* prow = pb + (w * 16 + row) * PSTR_;
        const int c0 = half * 136;
        float m = -1e30f;
        for (int c = c0; c < c0 + 136; ++c) m = fmaxf(m, sr[c]);
        m = fmaxf(m, __shfl_xor(m, 1, 32));
        float s = 0.f;
        for (int c = c0; c < c0 + 136; ++c) {
            const float e = __expf(sr[c] - m);
            prow[c] = f2bf(e);
            s += e;
        }
        s += __shfl_xor(s, 1, 32);
        for (int c = LK_ + half * 8; c < LK_ + half * 8 + 8; ++c)
            prow[c] = (unsigned short)0;
        if (half == 0) iv[w * 16 + row] = 1.0f / s;
    }
    __syncthreads();

    // ---- PV: out[16 x 128] = P[16 x 288] @ V[288 x 128], 9 k-steps ----
    v8f oacc[8];
    {
        const v8f z = {};
#pragma unroll
        for (int nt = 0; nt < 8; ++nt) oacc[nt] = z;
    }
    const unsigned short* pr = pb + (w * 16 + n16) * PSTR_;  // A lane row
    for (int ks = 0; ks < 9; ++ks) {
        Frag ap;
        frag_from_bf16lds(ap, pr + ks * 32, g);
#pragma unroll
        for (int nt = 0; nt < 8; ++nt) {
            Frag bV;   // V^T: column (d) per lane, contiguous along k
            frag_from_bf16lds(bV, vb + (nt * 16 + n16) * VSTR_ + ks * 32, g);
            oacc[nt] = __builtin_amdgcn_wmma_f32_16x16x32_bf16(false, ap.bf, false,
                                                               bV.bf, (short)0,
                                                               oacc[nt], false, false);
        }
    }

    // ---- fold 1/sum into store ----
#pragma unroll
    for (int nt = 0; nt < 8; ++nt) {
#pragma unroll
        for (int r = 0; r < 8; ++r) {
            const int M = r + 8 * g;
            const float sc = iv[w * 16 + M];
            out[(size_t)(q0 + M) * D_ + h * HD_ + nt * 16 + n16] = oacc[nt][r] * sc;
        }
    }
}

// ---------------------------------------------------------------------------
// Launcher
// inputs: q, ip_x, registers, Wk, bk, Wv, bv, norm_k_w, num_heads
// ---------------------------------------------------------------------------
extern "C" void kernel_launch(void* const* d_in, const int* in_sizes, int n_in,
                              void* d_out, int out_size, void* d_ws, size_t ws_size,
                              hipStream_t stream) {
    const float* q    = (const float*)d_in[0];
    const float* ipx  = (const float*)d_in[1];
    const float* regs = (const float*)d_in[2];
    const float* Wk   = (const float*)d_in[3];
    const float* bk   = (const float*)d_in[4];
    const float* Wv   = (const float*)d_in[5];
    const float* bv   = (const float*)d_in[6];
    const float* nkw  = (const float*)d_in[7];
    float*       out  = (float*)d_out;

    float* kraw = (float*)d_ws;                       // [272][5120] fp32
    float* vraw = kraw + (size_t)LK_ * D_;            // [272][5120] fp32

    // 1) K/V projections (WMMA bf16, fp32 accumulate)
    kv_proj_kernel<<<dim3(D_ / 128, LK_ / 16), 256, 0, stream>>>(
        ipx, regs, Wk, bk, Wv, bv, kraw, vraw);

    // 2) RMSNorm on K rows (in place, folds learnable weight)
    rmsnorm_k_kernel<<<LK_, 256, 0, stream>>>(kraw, nkw);

    // 3) Attention: 160 query chunks x 40 heads, ~297 KB LDS per WGP
    attn_kernel<<<dim3(S_ / 128, HEADS_), 256, SMEM_TOTAL, stream>>>(
        q, kraw, vraw, out);
}
